// LlamaPRMSearchForCausalLM_69904887710507
// MI455X (gfx1250) — matrix operand order repaired and checked by reading, not compiled, not run on hardware
//
#include <hip/hip_runtime.h>
#include <hip/hip_bf16.h>
#include <math.h>

#define B_      32
#define NBLK_   128
#define BLOCK_  16
#define H_      16
#define D_      128
#define HID_    2048
#define FFN_    8192
#define KVLEN_  2048

typedef __attribute__((ext_vector_type(16))) __bf16 v16bf;
typedef __attribute__((ext_vector_type(8)))  float  v8f;
typedef __attribute__((ext_vector_type(4)))  float  v4f;

// ln(10000) / (D/2) = 9.210340371976184 / 64
#define ROPE_LOGSTEP (9.210340371976184f / 64.0f)

// ---------------------------------------------------------------- embeddings
__global__ void embed_kernel(const int* __restrict__ ids,
                             const int* __restrict__ pos,
                             const float* __restrict__ tok,
                             const float* __restrict__ pemb,
                             float* __restrict__ emb) {
    int b = blockIdx.x;
    const v4f* t4 = (const v4f*)(tok  + (long)ids[b] * HID_);
    const v4f* p4 = (const v4f*)(pemb + (long)pos[b] * HID_);
    v4f* e4 = (v4f*)(emb + (long)b * HID_);
    for (int i = threadIdx.x; i < HID_ / 4; i += blockDim.x)
        e4[i] = t4[i] + p4[i];
}

// ------------------------------------------- skinny GEMM: C[32,N] = A[32,K]@W
// Split-K for HBM saturation: one block per 16-wide N tile; its 8 waves cover
// 2 M-tiles x 4 K-slices; partials reduced through LDS (deterministic order).
// K,N compile-time -> weight-row loads become immediate-offset b32 loads.
// Weights are streamed with non-temporal hint (read exactly once).
// Fragment layouts: contiguous k = 16*half + e in A and B (wave32, bf16 16x16x32).
template <int K, int N>
__global__ __launch_bounds__(256) void gemm32_bf16_wmma(
    const float* __restrict__ A, const float* __restrict__ W,
    const float* __restrict__ addsrc, float* __restrict__ C) {
    __shared__ float red[8][8][32];     // [r][wave][lane] : conflict-free

    int w    = threadIdx.x >> 5;        // wave 0..7
    int lane = threadIdx.x & 31;
    int tm   = w & 1;                   // M tile (M = 32)
    int ks   = w >> 1;                  // K slice 0..3
    int tn   = blockIdx.x;              // 16-wide N tile
    int half = lane >> 4;               // lane group 0..15 / 16..31
    int l16  = lane & 15;

    constexpr int kslice = K >> 2;      // K/4, divisible by 64
    int kbeg = ks * kslice;
    int kend = kbeg + kslice;

    const float* arowp = A + (long)(tm * 16 + l16) * K;
    int bcol = tn * 16 + l16;           // B fragment: N = lane%16

    v8f acc0 = {}, acc1 = {};
    for (int kc = kbeg; kc < kend; kc += 64) {
        #pragma unroll
        for (int u = 0; u < 2; ++u) {
            int k0 = kc + u * 32;
            // A fragment: elem e -> K = k0 + half*16 + e
            const v4f* ap4 = (const v4f*)(arowp + k0 + half * 16);  // 32B aligned
            v4f aa[4] = { ap4[0], ap4[1], ap4[2], ap4[3] };
            v16bf af;
            #pragma unroll
            for (int e = 0; e < 16; ++e) af[e] = (__bf16)aa[e >> 2][e & 3];

            // B fragment: elem e -> K = k0 + half*16 + e, N = bcol
            // constant row strides -> immediate-offset loads; NT (stream once)
            const float* wp = W + (long)(k0 + half * 16) * N + bcol;
            v16bf bf;
            #pragma unroll
            for (int e = 0; e < 16; ++e)
                bf[e] = (__bf16)__builtin_nontemporal_load(wp + (long)e * N);

            if (u == 0 && kc + 64 < kend)  // speculative prefetch, next chunk
                __builtin_prefetch(wp + (long)64 * N, 0, 1);

            if (u == 0)
                acc0 = __builtin_amdgcn_wmma_f32_16x16x32_bf16(
                           false, af, false, bf, (short)0, acc0, false, false);
            else
                acc1 = __builtin_amdgcn_wmma_f32_16x16x32_bf16(
                           false, af, false, bf, (short)0, acc1, false, false);
        }
    }

    #pragma unroll
    for (int r = 0; r < 8; ++r) red[r][w][lane] = acc0[r] + acc1[r];
    __syncthreads();

    // waves 0,1 (tm = w) sum the 4 K-slices and store
    if (w < 2) {
        if (addsrc) {
            #pragma unroll
            for (int r = 0; r < 8; ++r) {
                float v = red[r][w][lane]     + red[r][w + 2][lane]
                        + red[r][w + 4][lane] + red[r][w + 6][lane];
                long o = (long)(w * 16 + half * 8 + r) * N + tn * 16 + l16;
                C[o] = v + addsrc[o];
            }
        } else {
            #pragma unroll
            for (int r = 0; r < 8; ++r) {
                float v = red[r][w][lane]     + red[r][w + 2][lane]
                        + red[r][w + 4][lane] + red[r][w + 6][lane];
                long o = (long)(w * 16 + half * 8 + r) * N + tn * 16 + l16;
                C[o] = v;
            }
        }
    }
}

// ---------------------------------------------------------------- RoPE on q
__global__ void rope_q_kernel(float* __restrict__ q, const int* __restrict__ pos) {
    int idx = blockIdx.x * blockDim.x + threadIdx.x;  // B*H*64 threads
    int i  = idx & 63;
    int bh = idx >> 6;
    int b  = bh >> 4;
    long base = (long)bh * D_;
    float ang = (float)pos[b] * __expf(-(float)i * ROPE_LOGSTEP);
    float sn, cs; __sincosf(ang, &sn, &cs);
    float x1 = q[base + i], x2 = q[base + 64 + i];
    q[base + i]      = x1 * cs - x2 * sn;
    q[base + 64 + i] = x1 * sn + x2 * cs;
}

// --------------------------------------------------- paged decode attention
// One 256-thread block per (b,h). Inline RoPE on k at cache position.
// KV is 1.07 GB streamed exactly once -> non-temporal loads keep L2 clean.
__global__ __launch_bounds__(256) void attn_kernel(
    const float* __restrict__ q,     // [B][H][D] (roped)
    const float* __restrict__ kv,    // [NUM_BLOCKS][2][BLOCK][H][D]
    const int*  __restrict__ btab,   // [B][NBLK]
    const int*  __restrict__ ctxlen, // [B]
    float* __restrict__ out)         // [B][H*D]
{
    __shared__ float qs[D_];
    __shared__ float invfs[64];
    __shared__ float sc[KVLEN_];
    __shared__ float red[256];

    int b = blockIdx.x >> 4;
    int h = blockIdx.x & 15;
    int tid = threadIdx.x;
    int ctx = ctxlen[b];

    if (tid < D_) qs[tid] = q[(long)(b * H_ + h) * D_ + tid];
    if (tid < 64) invfs[tid] = __expf(-(float)tid * ROPE_LOGSTEP);
    __syncthreads();

    const float scale = 0.08838834764831845f; // 1/sqrt(128)

    // pass 1: scores with inline RoPE on k (float4 NT K reads)
    float lmax = -INFINITY;
    for (int t = tid; t < KVLEN_; t += 256) {
        float s_ = -1e30f;
        if (t < ctx) {
            int blk = btab[b * NBLK_ + (t >> 4)];
            const float* kp = kv + ((long)(blk * 2 + 0) * BLOCK_ + (t & 15)) * (H_ * D_) + h * D_;
            const v4f* kp4 = (const v4f*)kp;
            float dot = 0.f;
            #pragma unroll 2
            for (int i4 = 0; i4 < 16; ++i4) {
                v4f k1 = __builtin_nontemporal_load(kp4 + i4);
                v4f k2 = __builtin_nontemporal_load(kp4 + 16 + i4);
                #pragma unroll
                for (int j = 0; j < 4; ++j) {
                    int i = i4 * 4 + j;
                    float sn, cs; __sincosf((float)t * invfs[i], &sn, &cs);
                    dot += (k1[j] * cs - k2[j] * sn) * qs[i]
                         + (k1[j] * sn + k2[j] * cs) * qs[64 + i];
                }
            }
            s_ = dot * scale;
        }
        sc[t] = s_;
        lmax = fmaxf(lmax, s_);
    }
    red[tid] = lmax; __syncthreads();
    for (int o = 128; o > 0; o >>= 1) {
        if (tid < o) red[tid] = fmaxf(red[tid], red[tid + o]);
        __syncthreads();
    }
    float gmax = red[0]; __syncthreads();

    float lsum = 0.f;
    for (int t = tid; t < KVLEN_; t += 256) {
        float p = __expf(sc[t] - gmax);
        sc[t] = p;
        lsum += p;
    }
    red[tid] = lsum; __syncthreads();
    for (int o = 128; o > 0; o >>= 1) {
        if (tid < o) red[tid] += red[tid + o];
        __syncthreads();
    }
    float inv = 1.0f / red[0];
    __syncthreads();

    // pass 2: probs @ v — lane-coalesced: 128 consecutive dims per token,
    // two token-partitions summed via LDS.
    {
        int d = tid & 127;
        int part = tid >> 7;
        float acc = 0.f;
        for (int t = part; t < ctx; t += 2) {
            int blk = btab[b * NBLK_ + (t >> 4)];
            const float* vp = kv + ((long)(blk * 2 + 1) * BLOCK_ + (t & 15)) * (H_ * D_) + h * D_;
            acc += sc[t] * __builtin_nontemporal_load(vp + d);
        }
        red[tid] = acc;
    }
    __syncthreads();
    if (tid < D_)
        out[(long)(b * H_ + h) * D_ + tid] = (red[tid] + red[tid + 128]) * inv;
}

// ---------------------------------------------------------------------------
extern "C" void kernel_launch(void* const* d_in, const int* in_sizes, int n_in,
                              void* d_out, int out_size, void* d_ws, size_t ws_size,
                              hipStream_t stream) {
    (void)in_sizes; (void)n_in; (void)out_size; (void)ws_size;
    const int*   ids  = (const int*)  d_in[0];
    const int*   pos  = (const int*)  d_in[1];
    const float* kv   = (const float*)d_in[2];
    const int*   btab = (const int*)  d_in[3];
    const int*   ctx  = (const int*)  d_in[4];
    const float* etok = (const float*)d_in[5];
    const float* epos = (const float*)d_in[6];
    const float* Wq   = (const float*)d_in[7];
    const float* Wo   = (const float*)d_in[8];
    const float* Wprm = (const float*)d_in[9];
    const float* W1   = (const float*)d_in[10];
    const float* W2   = (const float*)d_in[11];
    float* out = (float*)d_out;

    float* ws     = (float*)d_ws;
    float* emb    = ws;                 // 32*2048
    float* q      = ws + 65536;         // 32*2048
    float* attn   = ws + 131072;        // 32*2048
    float* attnWo = ws + 196608;        // 32*2048
    float* prm    = ws + 262144;        // 32*2048
    float* hidden = ws + 327680;        // 32*8192

    embed_kernel<<<B_, 256, 0, stream>>>(ids, pos, etok, epos, emb);

    // split-K GEMMs: grid = N/16 blocks (128 for 2048-wide, 512 for FFN)
    gemm32_bf16_wmma<HID_, HID_><<<HID_ / 16, 256, 0, stream>>>(emb, Wq, nullptr, q);
    rope_q_kernel<<<(B_ * H_ * 64) / 256, 256, 0, stream>>>(q, pos);

    attn_kernel<<<B_ * H_, 256, 0, stream>>>(q, kv, btab, ctx, attn);

    gemm32_bf16_wmma<HID_, HID_><<<HID_ / 16, 256, 0, stream>>>(attn,   Wo,   nullptr, attnWo);
    gemm32_bf16_wmma<HID_, HID_><<<HID_ / 16, 256, 0, stream>>>(attnWo, Wprm, nullptr, prm);
    gemm32_bf16_wmma<HID_, FFN_><<<FFN_ / 16, 256, 0, stream>>>(attnWo, W1,   nullptr, hidden);
    // out = hidden @ W2 + prm   (fused residual add)
    gemm32_bf16_wmma<FFN_, HID_><<<HID_ / 16, 256, 0, stream>>>(hidden, W2, prm, out);
}
